// VNAttentionLayer_4724464025604
// MI455X (gfx1250) — matrix operand order repaired and checked by reading, not compiled, not run on hardware
//
#include <hip/hip_runtime.h>
#include <math.h>

// ---------------------------------------------------------------------------
// VN attention for MI455X (gfx1250, wave32, WMMA, async global->LDS copies).
// Heavy math via v_wmma_f32_16x16x32_bf16; data movement via
// global_load_async_to_lds_b128 (ASYNCcnt) where no conversion is needed,
// register-staged f32->bf16 pipeline (v_perm_b32-packed round-to-nearest)
// otherwise. Double-buffered LDS, 16B-aligned padded rows (bank-staggered).
// ---------------------------------------------------------------------------

typedef __attribute__((ext_vector_type(16))) __bf16       v16bf;
typedef __attribute__((ext_vector_type(8)))  float        v8f;
typedef __attribute__((ext_vector_type(8)))  unsigned int v8u;
typedef __attribute__((ext_vector_type(4)))  float        vf4;

#define BK   32   // K per WMMA step (bf16)
#define BKP  40   // padded LDS row stride (halves): 80B rows -> 16B aligned + 20-bank stagger

#define DMODEL 512
#define NHEADS 8
#define BB     4
#define LL     2048
#define EE     64              // head dim
#define FF     192             // 3 * EE : contracted feature dim per head
#define MROWS  (BB * LL * 3)   // 24576 rows for the big projections
#define WELEMS (DMODEL * DMODEL)  // 262144 elements per weight matrix

// f32 -> bf16 with round-to-nearest (half-up): +0x8000 then take high 16 bits.
// Packed pair via one v_perm_b32: 3 VALU per 2 elements.
__device__ __forceinline__ unsigned int f2bf_pk(float lo, float hi) {
  unsigned int ul = __builtin_bit_cast(unsigned int, lo) + 0x8000u;
  unsigned int uh = __builtin_bit_cast(unsigned int, hi) + 0x8000u;
  return __builtin_amdgcn_perm(uh, ul, 0x07060302u);  // {uh[31:16], ul[31:16]}
}
__device__ __forceinline__ unsigned short f2bf1(float f) {
  return (unsigned short)((__builtin_bit_cast(unsigned int, f) + 0x8000u) >> 16);
}

__device__ __forceinline__ v8f wmma_bf16(v8u a, v8u b, v8f c) {
  return __builtin_amdgcn_wmma_f32_16x16x32_bf16(
      false, __builtin_bit_cast(v16bf, a),
      false, __builtin_bit_cast(v16bf, b),
      (short)0, c, false, false);
}

// ---------------- CDNA5 async global->LDS (ASYNCcnt-tracked) ---------------
__device__ __forceinline__ void async_b128(unsigned lds_off, unsigned gvoff,
                                           const void* sbase) {
  // LDS byte offset from lds_off; global address is sbase plus byte offset gvoff
  asm volatile("global_load_async_to_lds_b128 %0, %1, %2"
               :: "v"(lds_off), "v"(gvoff), "s"(sbase) : "memory");
}
__device__ __forceinline__ void wait_async0() {
  asm volatile("s_wait_asynccnt 0x0" ::: "memory");
}

// Async copy of a ROWSx32 bf16 tile into padded LDS (16B per lane per issue).
template <int ROWS>
__device__ __forceinline__ void async_tile(unsigned lds_base, const unsigned short* src,
                                           unsigned ld, unsigned r0, unsigned k0, int tid) {
#pragma unroll
  for (int t = tid; t < ROWS * 4; t += 256) {
    unsigned r = t >> 2, c8 = (t & 3) << 3;
    unsigned gvoff = ((r0 + r) * ld + k0 + c8) * 2u;          // bytes
    unsigned loff  = lds_base + (r * BKP + c8) * 2u;          // bytes, 16B aligned
    async_b128(loff, gvoff, src);
  }
}

// ------------- register-staged f32 -> bf16 pipeline (convert path) ---------
template <int ROWS, bool NT>
__device__ __forceinline__ void fetch_f32(vf4 (&f)[ROWS / 32], const float* src,
                                          unsigned ld, unsigned r0, unsigned k0, int tid) {
#pragma unroll
  for (int j = 0; j < ROWS / 32; ++j) {
    int t = tid + j * 256;
    int r = t >> 3, c4 = (t & 7) << 2;
    const vf4* p = (const vf4*)(src + (size_t)(r0 + r) * ld + k0 + c4);
    f[j] = NT ? __builtin_nontemporal_load(p) : *p;
  }
}
template <int ROWS>
__device__ __forceinline__ void store_bf16(const vf4 (&f)[ROWS / 32],
                                           unsigned short* Sm, int tid) {
#pragma unroll
  for (int j = 0; j < ROWS / 32; ++j) {
    int t = tid + j * 256;
    int r = t >> 3, c4 = (t & 7) << 2;
    uint2* p = (uint2*)(Sm + r * BKP + c4);   // 8B aligned (80B rows, c4 mult of 4)
    uint2 d;
    d.x = f2bf_pk(f[j].x, f[j].y);
    d.y = f2bf_pk(f[j].z, f[j].w);
    *p = d;
  }
}

// ---------------- WMMA fragment loads (ISA 7.12.2 layouts) -----------------
__device__ __forceinline__ v8u load_a_frag(const unsigned short* As, int mrow, int half) {
  const unsigned short* base = As + mrow * BKP;
  v8u a;
#pragma unroll
  for (int i = 0; i < 8; ++i) {
    int kb = ((i & 4) << 2) + (half << 3) + ((i & 3) << 1);   // (i>=4?16:0)+8*half+2*(i&3)
    a[i] = *(const unsigned int*)(base + kb);
  }
  return a;
}
__device__ __forceinline__ v8u load_b_frag(const unsigned short* Bs, int nrow, int half) {
  const unsigned short* base = Bs + nrow * BKP;
  v8u b;
#pragma unroll
  for (int i = 0; i < 8; ++i) b[i] = *(const unsigned int*)(base + ((i & 4) << 2) + (half << 3) + ((i & 3) << 1));
  return b;
}

template <int TM, int TN>
__device__ __forceinline__ void mma_tiles(const unsigned short* As, const unsigned short* Bs,
                                          int wm, int wn, int l15, int half,
                                          v8f (&acc)[TM][TN]) {
  v8u af[TM], bf[TN];
#pragma unroll
  for (int i = 0; i < TM; ++i) af[i] = load_a_frag(As, wm + i * 16 + l15, half);
#pragma unroll
  for (int j = 0; j < TN; ++j) bf[j] = load_b_frag(Bs, wn + j * 16 + l15, half);
#pragma unroll
  for (int i = 0; i < TM; ++i)
#pragma unroll
    for (int j = 0; j < TN; ++j) acc[i][j] = wmma_bf16(af[i], bf[j], acc[i][j]);
}

__device__ __forceinline__ unsigned lds_off_u32(const void* p) {
  return (unsigned)(size_t)p;   // low 32 bits of generic ptr = LDS offset
}

// ---------------------------------------------------------------------------
// Kernel 0: convert the four 512x512 weight matrices to bf16 (kept [N][K]).
// ---------------------------------------------------------------------------
__global__ __launch_bounds__(256) void vn_w2bf_kernel(const float* __restrict__ W0,
                                                      const float* __restrict__ W1,
                                                      const float* __restrict__ W2,
                                                      const float* __restrict__ W3,
                                                      unsigned short* __restrict__ dst) {
  size_t i = ((size_t)blockIdx.x * 256 + threadIdx.x) * 4;   // over 4*WELEMS
  size_t m = i >> 18, off = i & (WELEMS - 1);
  const float* s = (m == 0) ? W0 : (m == 1) ? W1 : (m == 2) ? W2 : W3;
  vf4 f = *(const vf4*)(s + off);
  uint2 d;
  d.x = f2bf_pk(f.x, f.y);
  d.y = f2bf_pk(f.z, f.w);
  *(uint2*)(dst + i) = d;
}

// ---------------------------------------------------------------------------
// Kernel 1: projection Y = X @ W^T.  A: f32 register pipeline, B: async bf16.
// mode 0: dst[b][h][l][v*64+e]    mode 2: dst[b][h][v*64+e][l]  (V transposed)
// 128x128 block tile, 8 waves (2x4), wave tile 64x32.
// ---------------------------------------------------------------------------
__global__ __launch_bounds__(256) void vn_proj_kernel(const float* __restrict__ X,
                                                      const unsigned short* __restrict__ Wbf,
                                                      unsigned short* __restrict__ dst,
                                                      int mode) {
  __shared__ unsigned short As[128 * BKP];
  __shared__ unsigned short Bs[2][128 * BKP];
  const int tid = threadIdx.x, lane = tid & 31, wave = tid >> 5;
  const int half = lane >> 4, l15 = lane & 15;
  const int m0 = blockIdx.y * 128, n0 = blockIdx.x * 128;
  const int wm = (wave >> 2) * 64, wn = (wave & 3) * 32;
  const unsigned bbase[2] = {lds_off_u32(&Bs[0][0]), lds_off_u32(&Bs[1][0])};

  vf4 ra[4];
  fetch_f32<128, false>(ra, X, DMODEL, m0, 0, tid);
  async_tile<128>(bbase[0], Wbf, DMODEL, n0, 0, tid);

  v8f acc[4][2] = {};
  int cur = 0;
  for (int k0 = 0; k0 < DMODEL; k0 += BK) {
    __syncthreads();                       // everyone done reading As/Bs[cur^1]
    store_bf16<128>(ra, As, tid);
    wait_async0();                         // own async copies into Bs[cur] done
    __syncthreads();                       // all LDS data visible
    if (k0 + BK < DMODEL) {
      fetch_f32<128, false>(ra, X, DMODEL, m0, k0 + BK, tid);
      async_tile<128>(bbase[cur ^ 1], Wbf, DMODEL, n0, k0 + BK, tid);
    }
    mma_tiles<4, 2>(As, Bs[cur], wm, wn, l15, half, acc);
    cur ^= 1;
  }

#pragma unroll
  for (int ti = 0; ti < 4; ++ti)
#pragma unroll
    for (int tj = 0; tj < 2; ++tj) {
      int gn = n0 + wn + tj * 16 + l15;
      int h = gn >> 6, e = gn & 63;
#pragma unroll
      for (int i = 0; i < 8; ++i) {
        int gm = m0 + wm + ti * 16 + (half << 3) + i;
        int b = gm / (LL * 3), rem = gm % (LL * 3);
        int l = rem / 3, v = rem % 3;
        size_t idx = (mode == 2)
            ? (((size_t)b * NHEADS + h) * FF + v * EE + e) * LL + l
            : (((size_t)b * NHEADS + h) * LL + l) * FF + v * EE + e;
        dst[idx] = f2bf1(acc[ti][tj][i]);
      }
    }
}

// ---------------------------------------------------------------------------
// Kernel 2: scores S[bh][l][s] = 0.125 * sum_f Q'[l][f] K'[s][f]  (K=192).
// Both panels bf16 -> fully async, double-buffered.
// ---------------------------------------------------------------------------
__global__ __launch_bounds__(256) void vn_score_kernel(const unsigned short* __restrict__ Q,
                                                       const unsigned short* __restrict__ Kp,
                                                       float* __restrict__ attn) {
  __shared__ unsigned short As[2][128 * BKP];
  __shared__ unsigned short Bs[2][128 * BKP];
  const int tid = threadIdx.x, lane = tid & 31, wave = tid >> 5;
  const int half = lane >> 4, l15 = lane & 15;
  const int bh = blockIdx.z;
  const unsigned short* Qb = Q  + (size_t)bh * LL * FF;
  const unsigned short* Kb = Kp + (size_t)bh * LL * FF;
  const int m0 = blockIdx.y * 128, n0 = blockIdx.x * 128;
  const int wm = (wave >> 2) * 64, wn = (wave & 3) * 32;
  const unsigned abase[2] = {lds_off_u32(&As[0][0]), lds_off_u32(&As[1][0])};
  const unsigned bbase[2] = {lds_off_u32(&Bs[0][0]), lds_off_u32(&Bs[1][0])};

  async_tile<128>(abase[0], Qb, FF, m0, 0, tid);
  async_tile<128>(bbase[0], Kb, FF, n0, 0, tid);

  v8f acc[4][2] = {};
  int cur = 0;
  for (int k0 = 0; k0 < FF; k0 += BK) {    // 6 steps
    wait_async0();
    __syncthreads();
    if (k0 + BK < FF) {
      async_tile<128>(abase[cur ^ 1], Qb, FF, m0, k0 + BK, tid);
      async_tile<128>(bbase[cur ^ 1], Kb, FF, n0, k0 + BK, tid);
    }
    mma_tiles<4, 2>(As[cur], Bs[cur], wm, wn, l15, half, acc);
    cur ^= 1;
  }

  float* Ab = attn + (size_t)bh * LL * LL;
#pragma unroll
  for (int ti = 0; ti < 4; ++ti)
#pragma unroll
    for (int tj = 0; tj < 2; ++tj) {
      int gn = n0 + wn + tj * 16 + l15;
#pragma unroll
      for (int i = 0; i < 8; ++i) {
        int gm = m0 + wm + ti * 16 + (half << 3) + i;
        Ab[(size_t)gm * LL + gn] = 0.125f * acc[ti][tj][i];
      }
    }
}

// ---------------------------------------------------------------------------
// Kernel 3: in-place row softmax, rows of 2048 (vectorized 512B per wavefront).
// ---------------------------------------------------------------------------
__global__ __launch_bounds__(256) void vn_softmax_kernel(float* __restrict__ attn) {
  __shared__ float red[256];
  const int tid = threadIdx.x;
  vf4* p = (vf4*)(attn + (size_t)blockIdx.x * LL);

  vf4 a = p[tid], b = p[tid + 256];
  float m = fmaxf(fmaxf(fmaxf(a.x, a.y), fmaxf(a.z, a.w)),
                  fmaxf(fmaxf(b.x, b.y), fmaxf(b.z, b.w)));
  red[tid] = m; __syncthreads();
  for (int s = 128; s > 0; s >>= 1) { if (tid < s) red[tid] = fmaxf(red[tid], red[tid + s]); __syncthreads(); }
  m = red[0]; __syncthreads();

  a.x = __expf(a.x - m); a.y = __expf(a.y - m); a.z = __expf(a.z - m); a.w = __expf(a.w - m);
  b.x = __expf(b.x - m); b.y = __expf(b.y - m); b.z = __expf(b.z - m); b.w = __expf(b.w - m);
  float sum = (a.x + a.y) + (a.z + a.w) + (b.x + b.y) + (b.z + b.w);
  red[tid] = sum; __syncthreads();
  for (int s = 128; s > 0; s >>= 1) { if (tid < s) red[tid] += red[tid + s]; __syncthreads(); }
  float inv = 1.0f / red[0];
  a *= inv; b *= inv;
  p[tid] = a; p[tid + 256] = b;
}

// ---------------------------------------------------------------------------
// Kernel 4: O'[l][f] = sum_s attn[l][s] V'[s][f]  (K=2048, N=192).
// A: f32 attn, NT reads (537MB single-use stream > L2), register pipeline.
// B: async bf16 from transposed V. 128x64 tile, 8 waves (4x2), wave 32x32.
// ---------------------------------------------------------------------------
__global__ __launch_bounds__(256) void vn_av_kernel(const float* __restrict__ attn,
                                                    const unsigned short* __restrict__ Vt,
                                                    unsigned short* __restrict__ out_pre) {
  __shared__ unsigned short As[128 * BKP];
  __shared__ unsigned short Bs[2][64 * BKP];
  const int tid = threadIdx.x, lane = tid & 31, wave = tid >> 5;
  const int half = lane >> 4, l15 = lane & 15;
  const int bh = blockIdx.z;
  const int b = bh >> 3, h = bh & 7;
  const float* Ab = attn + (size_t)bh * LL * LL;
  const unsigned short* Vb = Vt + (size_t)bh * FF * LL;     // [f][s]
  const int m0 = blockIdx.y * 128, n0 = blockIdx.x * 64;
  const int wm = (wave >> 1) * 32, wn = (wave & 1) * 32;
  const unsigned bbase[2] = {lds_off_u32(&Bs[0][0]), lds_off_u32(&Bs[1][0])};

  vf4 ra[4];
  fetch_f32<128, true>(ra, Ab, LL, m0, 0, tid);
  async_tile<64>(bbase[0], Vb, LL, n0, 0, tid);

  v8f acc[2][2] = {};
  int cur = 0;
  for (int k0 = 0; k0 < LL; k0 += BK) {    // 64 steps
    __syncthreads();
    store_bf16<128>(ra, As, tid);
    wait_async0();
    __syncthreads();
    if (k0 + BK < LL) {
      fetch_f32<128, true>(ra, Ab, LL, m0, k0 + BK, tid);
      async_tile<64>(bbase[cur ^ 1], Vb, LL, n0, k0 + BK, tid);
    }
    mma_tiles<2, 2>(As, Bs[cur], wm, wn, l15, half, acc);
    cur ^= 1;
  }

#pragma unroll
  for (int ti = 0; ti < 2; ++ti)
#pragma unroll
    for (int tj = 0; tj < 2; ++tj) {
      int gn = n0 + wn + tj * 16 + l15;    // f in [0,192)
      int v = gn / EE, e = gn % EE;
#pragma unroll
      for (int i = 0; i < 8; ++i) {
        int gm = m0 + wm + ti * 16 + (half << 3) + i;    // l
        size_t idx = (((size_t)b * LL + gm) * 3 + v) * DMODEL + h * EE + e;
        out_pre[idx] = f2bf1(acc[ti][tj][i]);
      }
    }
}

// ---------------------------------------------------------------------------
// Kernel 5: out = out_pre @ Wo^T (f32 result). Both panels bf16 -> fully async.
// ---------------------------------------------------------------------------
__global__ __launch_bounds__(256) void vn_out_kernel(const unsigned short* __restrict__ Xp,
                                                     const unsigned short* __restrict__ Wobf,
                                                     float* __restrict__ out) {
  __shared__ unsigned short As[2][128 * BKP];
  __shared__ unsigned short Bs[2][128 * BKP];
  const int tid = threadIdx.x, lane = tid & 31, wave = tid >> 5;
  const int half = lane >> 4, l15 = lane & 15;
  const int m0 = blockIdx.y * 128, n0 = blockIdx.x * 128;
  const int wm = (wave >> 2) * 64, wn = (wave & 3) * 32;
  const unsigned abase[2] = {lds_off_u32(&As[0][0]), lds_off_u32(&As[1][0])};
  const unsigned bbase[2] = {lds_off_u32(&Bs[0][0]), lds_off_u32(&Bs[1][0])};

  async_tile<128>(abase[0], Xp, DMODEL, m0, 0, tid);
  async_tile<128>(bbase[0], Wobf, DMODEL, n0, 0, tid);

  v8f acc[4][2] = {};
  int cur = 0;
  for (int k0 = 0; k0 < DMODEL; k0 += BK) {   // 16 steps
    wait_async0();
    __syncthreads();
    if (k0 + BK < DMODEL) {
      async_tile<128>(abase[cur ^ 1], Xp, DMODEL, m0, k0 + BK, tid);
      async_tile<128>(bbase[cur ^ 1], Wobf, DMODEL, n0, k0 + BK, tid);
    }
    mma_tiles<4, 2>(As[cur], Bs[cur], wm, wn, l15, half, acc);
    cur ^= 1;
  }

#pragma unroll
  for (int ti = 0; ti < 4; ++ti)
#pragma unroll
    for (int tj = 0; tj < 2; ++tj) {
      int gn = n0 + wn + tj * 16 + l15;
#pragma unroll
      for (int i = 0; i < 8; ++i) {
        int gm = m0 + wm + ti * 16 + (half << 3) + i;
        out[(size_t)gm * DMODEL + gn] = acc[ti][tj][i];
      }
    }
}

// ---------------------------------------------------------------------------
extern "C" void kernel_launch(void* const* d_in, const int* in_sizes, int n_in,
                              void* d_out, int out_size, void* d_ws, size_t ws_size,
                              hipStream_t stream) {
  (void)in_sizes; (void)n_in; (void)out_size; (void)ws_size;
  const float* queries = (const float*)d_in[0];
  const float* keys    = (const float*)d_in[1];
  const float* values  = (const float*)d_in[2];
  const float* Wq      = (const float*)d_in[3];
  const float* Wk      = (const float*)d_in[4];
  const float* Wv      = (const float*)d_in[5];
  const float* Wo      = (const float*)d_in[6];

  float* out  = (float*)d_out;                              // [B,L,3,512]
  float* attn = out + (size_t)BB * LL * 3 * DMODEL;         // [B,H,L,L]

  const size_t PROJ_ELEMS = (size_t)BB * NHEADS * LL * FF;  // 12,582,912 bf16 each
  unsigned short* ws    = (unsigned short*)d_ws;
  unsigned short* q_ws  = ws;                      // Q'  [bh][l][f]
  unsigned short* k_ws  = ws + PROJ_ELEMS;         // K'  [bh][s][f]
  unsigned short* vt_ws = ws + 2 * PROJ_ELEMS;     // V'^T [bh][f][s]
  unsigned short* w_ws  = ws + 3 * PROJ_ELEMS;     // Wq|Wk|Wv|Wo bf16, [N][K]
  unsigned short* op_ws = q_ws;                    // out_pre aliases Q (dead after scores)

  vn_w2bf_kernel<<<dim3(4 * WELEMS / 1024), 256, 0, stream>>>(Wq, Wk, Wv, Wo, w_ws);

  dim3 gproj(DMODEL / 128, MROWS / 128);           // (4, 192)
  vn_proj_kernel<<<gproj, 256, 0, stream>>>(queries, w_ws + 0 * WELEMS, q_ws, 0);
  vn_proj_kernel<<<gproj, 256, 0, stream>>>(keys,    w_ws + 1 * WELEMS, k_ws, 0);
  vn_proj_kernel<<<gproj, 256, 0, stream>>>(values,  w_ws + 2 * WELEMS, vt_ws, 2);

  vn_score_kernel<<<dim3(LL / 128, LL / 128, BB * NHEADS), 256, 0, stream>>>(q_ws, k_ws, attn);
  vn_softmax_kernel<<<dim3(BB * NHEADS * LL), 256, 0, stream>>>(attn);
  vn_av_kernel<<<dim3(FF / 64, LL / 128, BB * NHEADS), 256, 0, stream>>>(attn, vt_ws, op_ws);
  vn_out_kernel<<<dim3(DMODEL / 128, MROWS / 128), 256, 0, stream>>>(op_ws, w_ws + 3 * WELEMS, out);
}
